// TransitionDown_19610820674066
// MI455X (gfx1250) — matrix operand order repaired and checked by reading, not compiled, not run on hardware
//
#include <hip/hip_runtime.h>
#include <stdint.h>

#define B_   4
#define N_   16384
#define D_   125
#define C0_  128
#define C1_  128
#define C2_  256
#define S_   2048
#define K_   16
#define M_   (B_*S_*K_)   // 131072 pixels
#define QTOT_ (B_*S_)     // 8192 queries

typedef __attribute__((ext_vector_type(16))) __bf16 v16bf;
typedef __attribute__((ext_vector_type(8)))  float  v8f;

static __device__ __forceinline__ unsigned short f2bf(float f) {
  unsigned u = __float_as_uint(f);
  u += 0x7FFFu + ((u >> 16) & 1u);   // round-to-nearest-even
  return (unsigned short)(u >> 16);
}

// ---------------------------------------------------------------------------
// Prep: convert W1 (128x128) and W2 (256x128) to bf16 (row-major, o-major).
// ---------------------------------------------------------------------------
__global__ void k_prep(const float* __restrict__ W1, const float* __restrict__ W2,
                       unsigned short* __restrict__ W1b, unsigned short* __restrict__ W2b) {
  int i = blockIdx.x * 256 + threadIdx.x;
  if (i < C1_ * C0_) W1b[i] = f2bf(W1[i]);
  if (i < C2_ * C1_) W2b[i] = f2bf(W2[i]);
}

// ---------------------------------------------------------------------------
// FPS: one workgroup per batch, point cloud register-resident (16 pts/thread),
// LDS tree argmax each step. Emits new_xyz[b][s] = xyz[b][farthest].
// ---------------------------------------------------------------------------
__global__ void __launch_bounds__(1024) k_fps(const float* __restrict__ xyz,
                                              float* __restrict__ newxyz) {
  const int b = blockIdx.x;
  const int tid = threadIdx.x;
  const float* xb = xyz + (size_t)b * N_ * 3;

  float px[16], py[16], pz[16], dmin[16];
#pragma unroll
  for (int j = 0; j < 16; ++j) {
    int p = j * 1024 + tid;
    px[j] = xb[p * 3 + 0];
    py[j] = xb[p * 3 + 1];
    pz[j] = xb[p * 3 + 2];
    dmin[j] = 1e10f;
  }

  __shared__ float sval[1024];
  __shared__ int   sidx[1024];

  int f = 0;
  for (int s = 0; s < S_; ++s) {
    float cx = xb[3 * f + 0], cy = xb[3 * f + 1], cz = xb[3 * f + 2];
    if (tid == 0) {
      float* o = newxyz + ((size_t)b * S_ + s) * 3;
      o[0] = cx; o[1] = cy; o[2] = cz;
    }
    float best = -1.0f; int bi = 0;
#pragma unroll
    for (int j = 0; j < 16; ++j) {
      float dx = px[j] - cx, dy = py[j] - cy, dz = pz[j] - cz;
      float d = dx * dx + dy * dy + dz * dz;
      float dm = fminf(dmin[j], d);
      dmin[j] = dm;
      if (dm > best) { best = dm; bi = j * 1024 + tid; }
    }
    sval[tid] = best; sidx[tid] = bi;
    __syncthreads();
    for (int off = 512; off > 0; off >>= 1) {
      if (tid < off) {
        float v2 = sval[tid + off]; int i2 = sidx[tid + off];
        float v1 = sval[tid];       int i1 = sidx[tid];
        if (v2 > v1 || (v2 == v1 && i2 < i1)) { sval[tid] = v2; sidx[tid] = i2; }
      }
      __syncthreads();
    }
    f = sidx[0];
    __syncthreads();
  }
}

// ---------------------------------------------------------------------------
// kNN: 1 thread per query; xyz chunked through LDS (broadcast reads),
// register-resident top-16 via guarded insertion (ties keep earlier index).
// ---------------------------------------------------------------------------
#define CH_ 2048
__global__ void k_knn(const float* __restrict__ xyz, const float* __restrict__ newxyz,
                      int* __restrict__ knn) {
  const int b = blockIdx.y;
  const int q = blockIdx.x * 128 + threadIdx.x;
  const int tid = threadIdx.x;
  const float* xb = xyz + (size_t)b * N_ * 3;
  const float* nq = newxyz + ((size_t)b * S_ + q) * 3;
  float qx = nq[0], qy = nq[1], qz = nq[2];

  float bd[K_]; int bi[K_];
#pragma unroll
  for (int j = 0; j < K_; ++j) { bd[j] = 3.0e38f; bi[j] = 0; }

  __shared__ float sx[CH_], sy[CH_], sz[CH_];
  for (int c0 = 0; c0 < N_; c0 += CH_) {
    for (int t = tid; t < CH_; t += 128) {
      const float* pp = xb + (size_t)(c0 + t) * 3;
      sx[t] = pp[0]; sy[t] = pp[1]; sz[t] = pp[2];
    }
    __syncthreads();
    for (int t = 0; t < CH_; ++t) {
      float dx = sx[t] - qx, dy = sy[t] - qy, dz = sz[t] - qz;
      float d = dx * dx + dy * dy + dz * dz;
      if (d < bd[K_ - 1]) {
        float cd = d; int ci = c0 + t;
#pragma unroll
        for (int j = 0; j < K_; ++j) {
          if (cd < bd[j]) {
            float td = bd[j]; int ti = bi[j];
            bd[j] = cd; bi[j] = ci; cd = td; ci = ti;
          }
        }
      }
    }
    __syncthreads();
  }
  int* o = knn + ((size_t)b * S_ + q) * K_;
#pragma unroll
  for (int j = 0; j < K_; ++j) o[j] = bi[j];
}

// ---------------------------------------------------------------------------
// Gather + concat + bf16 convert: Xin[p][c], p=(b,s,kn), c = 3 xyz_norm + 125 feat.
// One block per pixel, one thread per channel -> coalesced writes.
// ---------------------------------------------------------------------------
__global__ void k_gather(const float* __restrict__ xyz, const float* __restrict__ feat,
                         const float* __restrict__ newxyz, const int* __restrict__ knn,
                         unsigned short* __restrict__ Xb) {
  int p = blockIdx.x;
  int c = threadIdx.x;           // 0..127
  int b = p >> 15;               // / (S_*K_) == 32768
  int rem = p & 32767;
  int s = rem >> 4;
  int pt = knn[p];
  float v;
  if (c < 3) {
    v = xyz[((size_t)b * N_ + pt) * 3 + c] - newxyz[((size_t)b * S_ + s) * 3 + c];
  } else {
    v = feat[((size_t)b * N_ + pt) * D_ + (c - 3)];
  }
  Xb[(size_t)p * C0_ + c] = f2bf(v);
}

// ---------------------------------------------------------------------------
// bf16 WMMA GEMM: Y[c][p] = sum_k W[c][k] * X[p][k].  K = 128 fixed, M = 131072.
// Block: 64-pixel tile x COUT channels, (COUT/16) waves (blockDim = COUT*2).
// B tile (64 px x 128 ch bf16 = 16KB) staged in LDS via the CDNA5 async path
// (global_load_async_to_lds_b128 + s_wait_asynccnt), shared by all waves.
// Each wave preloads its 4 A (weight) K-step operands once into registers,
// then runs 4 pixel-subtiles x 4 K-steps = 16 back-to-back WMMAs from LDS.
// Register layouts used here (contiguous k, same order for A and B):
//   A 16x32: lane m=lane&15; K in {0..15} (lanes<16) / {16..31}
//   B 32x16: lane n=lane&15; K in {0..15} (lanes<16) / {16..31}
// ---------------------------------------------------------------------------
template<int COUT>
__global__ void k_gemm(const unsigned short* __restrict__ Xb,
                       const unsigned short* __restrict__ Wb,
                       float* __restrict__ Y) {
  __shared__ __align__(16) unsigned short sB[64 * C0_];   // 16KB
  const int tid = threadIdx.x;
  const int p0 = blockIdx.x * 64;

  // --- async stage of the B tile: 1024 x 16B chunks, spread over the block ---
  {
    const char* gbase = (const char*)(Xb + (size_t)p0 * C0_);
    unsigned ldsbase = (unsigned)(uintptr_t)(&sB[0]);   // low 32 bits = LDS offset
    for (int i = tid; i < 1024; i += COUT * 2) {
      unsigned loff = ldsbase + (unsigned)i * 16u;
      unsigned long long g = (unsigned long long)(uintptr_t)(gbase + (size_t)i * 16u);
      asm volatile("global_load_async_to_lds_b128 %0, %1, off"
                   :: "v"(loff), "v"(g) : "memory");
    }
    asm volatile("s_wait_asynccnt 0x0" ::: "memory");
    __syncthreads();
  }

  const int wave = tid >> 5;
  const int lane = tid & 31;
  const int o0  = wave * 16;
  const int row = o0 + (lane & 15);
  const int col = lane & 15;
  const int kbA = (lane < 16) ? 0 : 16;
  const int kbB = (lane < 16) ? 0 : 16;

  union U { unsigned u32[8]; v16bf v; };

  // Preload all A operands (weights) for the 4 K-steps: 32 VGPRs.
  U A[4];
#pragma unroll
  for (int ks = 0; ks < 4; ++ks) {
#pragma unroll
    for (int vv = 0; vv < 8; ++vv) {
      int koffA = kbA + 2 * vv;
      A[ks].u32[vv] = *(const unsigned*)(Wb + (size_t)row * C0_ + ks * 32 + koffA);
    }
  }

#pragma unroll
  for (int pt = 0; pt < 4; ++pt) {
    const unsigned short* sp = sB + (pt * 16 + col) * C0_;
    v8f acc = {0.f, 0.f, 0.f, 0.f, 0.f, 0.f, 0.f, 0.f};
#pragma unroll
    for (int ks = 0; ks < 4; ++ks) {
      U Bv;
#pragma unroll
      for (int vv = 0; vv < 8; ++vv) {
        Bv.u32[vv] = *(const unsigned*)(sp + ks * 32 + kbB + 2 * vv);
      }
      acc = __builtin_amdgcn_wmma_f32_16x16x32_bf16(
          /*neg_a=*/false, A[ks].v, /*neg_b=*/false, Bv.v,
          /*c_mod=*/(short)0, acc, /*reuse_a=*/false, /*reuse_b=*/false);
    }
    // Store D tile: base + constant row offsets (r*M_*4 fits signed 24-bit IOFFSET)
    const int p = p0 + pt * 16 + (lane & 15);
    float* yb = Y + (size_t)(o0 + ((lane >> 4) << 3)) * M_ + p;
#pragma unroll
    for (int r = 0; r < 8; ++r) {
      yb[(size_t)r * M_] = acc[r];
    }
  }
}

// ---------------------------------------------------------------------------
// Per-channel batch stats (biased var) -> fold BN affine into (scale, shift).
// One block per channel; coalesced reads of Y[c][*].
// ---------------------------------------------------------------------------
__global__ void k_stats(const float* __restrict__ Y, const float* __restrict__ gamma,
                        const float* __restrict__ beta, float* __restrict__ scale,
                        float* __restrict__ shift) {
  __shared__ float ss[256], ss2[256];
  int c = blockIdx.x, tid = threadIdx.x;
  const float* rowp = Y + (size_t)c * M_;
  float s = 0.f, s2 = 0.f;
  for (int i = tid; i < M_; i += 256) { float v = rowp[i]; s += v; s2 += v * v; }
  ss[tid] = s; ss2[tid] = s2;
  __syncthreads();
  for (int off = 128; off > 0; off >>= 1) {
    if (tid < off) { ss[tid] += ss[tid + off]; ss2[tid] += ss2[tid + off]; }
    __syncthreads();
  }
  if (tid == 0) {
    float mean = ss[0] / (float)M_;
    float var  = ss2[0] / (float)M_ - mean * mean;
    float rstd = rsqrtf(var + 1e-5f);
    float a = rstd * gamma[c];
    scale[c] = a;
    shift[c] = beta[c] - mean * a;
  }
}

// ---------------------------------------------------------------------------
// BN1 + ReLU + transpose to pixel-major bf16 for GEMM2 B operand.
// ---------------------------------------------------------------------------
__global__ void k_bnact(const float* __restrict__ Y, const float* __restrict__ scale,
                        const float* __restrict__ shift, unsigned short* __restrict__ out) {
  int p = blockIdx.x * 256 + threadIdx.x;
  int c = blockIdx.y;
  float v = Y[(size_t)c * M_ + p];
  v = fmaxf(fmaf(scale[c], v, shift[c]), 0.f);
  out[(size_t)p * C1_ + c] = f2bf(v);
}

// ---------------------------------------------------------------------------
// BN2 + ReLU + max over k neighbors + transpose -> x[B,S,C2].
// ---------------------------------------------------------------------------
__global__ void k_out(const float* __restrict__ Y2, const float* __restrict__ scale,
                      const float* __restrict__ shift, float* __restrict__ xout) {
  int q = blockIdx.x * 256 + threadIdx.x;  // 0..QTOT_-1
  int c = blockIdx.y;                      // 0..C2_-1
  float a = scale[c], b = shift[c];
  const float* base = Y2 + (size_t)c * M_ + (size_t)q * 16;
  float m = -3.4e38f;
#pragma unroll
  for (int kn = 0; kn < 16; ++kn) {
    float t = fmaxf(fmaf(a, base[kn], b), 0.f);
    m = fmaxf(m, t);
  }
  xout[(size_t)q * C2_ + c] = m;
}

// ---------------------------------------------------------------------------
extern "C" void kernel_launch(void* const* d_in, const int* in_sizes, int n_in,
                              void* d_out, int out_size, void* d_ws, size_t ws_size,
                              hipStream_t stream) {
  const float* xyz  = (const float*)d_in[0];
  const float* feat = (const float*)d_in[1];
  const float* W1   = (const float*)d_in[2];
  const float* g1   = (const float*)d_in[4];
  const float* be1  = (const float*)d_in[5];
  const float* W2   = (const float*)d_in[6];
  const float* g2   = (const float*)d_in[8];
  const float* be2  = (const float*)d_in[9];

  float* out_newxyz = (float*)d_out;                         // [B,S,3]
  float* out_x      = (float*)d_out + (size_t)B_ * S_ * 3;   // [B,S,C2]

  char* ws = (char*)d_ws;
  size_t off = 0;
  auto alloc = [&](size_t bytes) -> char* {
    char* p = ws + off;
    off = (off + bytes + 255) & ~(size_t)255;
    return p;
  };
  int*            knn = (int*)           alloc(sizeof(int) * (size_t)M_);
  unsigned short* W1b = (unsigned short*)alloc(2ull * C1_ * C0_);
  unsigned short* W2b = (unsigned short*)alloc(2ull * C2_ * C1_);
  float*          sc1 = (float*)         alloc(4ull * C1_);
  float*          sh1 = (float*)         alloc(4ull * C1_);
  float*          sc2 = (float*)         alloc(4ull * C2_);
  float*          sh2 = (float*)         alloc(4ull * C2_);
  unsigned short* Xb  = (unsigned short*)alloc(2ull * M_ * C0_);
  float*          Y1  = (float*)         alloc(4ull * C1_ * M_);
  unsigned short* X1b = (unsigned short*)alloc(2ull * M_ * C1_);
  float*          Y2  = (float*)         alloc(4ull * C2_ * M_);

  k_prep  <<<(C2_ * C1_ + 255) / 256, 256, 0, stream>>>(W1, W2, W1b, W2b);
  k_fps   <<<B_, 1024, 0, stream>>>(xyz, out_newxyz);
  k_knn   <<<dim3(S_ / 128, B_), 128, 0, stream>>>(xyz, out_newxyz, knn);
  k_gather<<<M_, 128, 0, stream>>>(xyz, feat, out_newxyz, knn, Xb);

  k_gemm<C1_><<<M_ / 64, C1_ * 2, 0, stream>>>(Xb, W1b, Y1);
  k_stats    <<<C1_, 256, 0, stream>>>(Y1, g1, be1, sc1, sh1);
  k_bnact    <<<dim3(M_ / 256, C1_), 256, 0, stream>>>(Y1, sc1, sh1, X1b);

  k_gemm<C2_><<<M_ / 64, C2_ * 2, 0, stream>>>(X1b, W2b, Y2);
  k_stats    <<<C2_, 256, 0, stream>>>(Y2, g2, be2, sc2, sh2);
  k_out      <<<dim3(QTOT_ / 256, C2_), 256, 0, stream>>>(Y2, sc2, sh2, out_x);
}
